// MyRnn_76725295776088
// MI455X (gfx1250) — matrix operand order repaired and checked by reading, not compiled, not run on hardware
//
#include <hip/hip_runtime.h>
#include <hip/hip_bf16.h>

// ---------------------------------------------------------------------------
// SimpleRNN on MI455X (gfx1250, wave32):
//   xp = x @ Wx + b          : bf16 WMMA GEMM (32768 x 512 x 1024), Wx slice
//                              TDM-staged into LDS per block
//   h_t = tanh(xp_t + h@Wh)  : persistent kernel; Wh slice TDM-staged into
//                              LDS once, resident for all 512 timesteps;
//                              grid barrier per timestep
// ---------------------------------------------------------------------------

typedef __attribute__((ext_vector_type(16))) __bf16 v16bf;
typedef __attribute__((ext_vector_type(8)))  float  v8f;

typedef __attribute__((ext_vector_type(4))) unsigned tdm_v4u;
typedef __attribute__((ext_vector_type(8))) int      tdm_v8i;
typedef __attribute__((ext_vector_type(4))) int      tdm_v4i;

union FragU {
    uint4          q[2];
    unsigned short us[16];
    __bf16         bh[16];
    v16bf          v;
};

__device__ __forceinline__ unsigned short f2bf(float f) {
    return __builtin_bit_cast(unsigned short, (__bf16)f);
}
__device__ __forceinline__ float bf2f(unsigned short s) {
    return (float)__builtin_bit_cast(__bf16, s);
}

#define B_  64
#define T_  512
#define D_  512
#define U_  1024
#define NWG_RNN 64

// ---------------------------------------------------------------------------
// Tensor Data Mover: load a [rows x 16] bf16 tile (column slice of a
// row-major [*, 1024] bf16 matrix) from global into LDS at lds_off.
// D# packing per CDNA5 ISA (2D tensor, groups 2/3 zero):
//   g0: count=1 | lds_addr | global_addr lo | global_addr hi + type=2
//   g1: data_size=2B | tensor_dim0=1024 | tensor_dim1=rows | tile_dim0=16 |
//       tile_dim1=rows | tensor_dim0_stride=1024
// ---------------------------------------------------------------------------
__device__ __forceinline__ void tdm_load_col_tile(unsigned lds_off,
                                                  const unsigned short* gbase,
                                                  unsigned rows) {
    const unsigned long long ga = (unsigned long long)(uintptr_t)gbase;
    tdm_v4u g0 = { 1u,                                   // count=1, user mode
                   lds_off,                              // LDS byte address
                   (unsigned)(ga & 0xFFFFFFFFu),         // global_addr[31:0]
                   (unsigned)(((ga >> 32) & 0x1FFFFFFu) | 0x80000000u) }; // +type=2
    tdm_v8i g1 = { 0x00010000,                           // data_size=2 bytes
                   (int)((U_ & 0xFFFFu) << 16),          // tensor_dim0 lo16
                   (int)((rows & 0xFFFFu) << 16),        // dim0 hi=0 | tensor_dim1 lo16
                   (int)(16u << 16),                     // dim1 hi=0 | tile_dim0=16
                   (int)(rows & 0xFFFFu),                // tile_dim1=rows | tile_dim2=0
                   (int)U_,                              // tensor_dim0_stride lo32
                   0, 0 };                               // stride0 hi | stride1
    tdm_v4i gz = { 0, 0, 0, 0 };
#if __has_include(<hip/amd_detail/amd_gfx1250_TDM.h>)
    tdm_v8i gz8 = { 0, 0, 0, 0, 0, 0, 0, 0 };
    __builtin_amdgcn_tensor_load_to_lds(g0, g1, gz, gz, gz8, 0);
#else
    __builtin_amdgcn_tensor_load_to_lds(g0, g1, gz, gz, 0);
#endif
}

// ---------------------------------------------------------------------------
// Init: zero h0 (bf16 [B][U]) and the two barrier counters.
// ---------------------------------------------------------------------------
__global__ void rnn_init_kernel(unsigned* __restrict__ h0w, unsigned* __restrict__ ctr) {
    const int idx = blockIdx.x * blockDim.x + threadIdx.x;
    if (idx < (B_ * U_ / 2)) h0w[idx] = 0u;   // 32768 dwords
    if (idx == 0) { ctr[0] = 0u; ctr[1] = 0u; }
}

// ---------------------------------------------------------------------------
// Convert fp32 -> bf16 (grid-stride, coalesced). Used for Wx and Wh.
// ---------------------------------------------------------------------------
__global__ void cvt_bf16_kernel(const float* __restrict__ src,
                                unsigned short* __restrict__ dst, int n) {
    for (int j = blockIdx.x * blockDim.x + threadIdx.x; j < n;
         j += gridDim.x * blockDim.x)
        dst[j] = f2bf(src[j]);
}

// ---------------------------------------------------------------------------
// xp GEMM: block owns columns [u0,u0+16); 8 waves split 128 rows (16 each).
// Wx column slice (512x16 bf16, 16 KB) is TDM-loaded into LDS raw, then
// swizzled once into B-fragment order: WxL[(kb*32+lane)*16 + e] with
// K = kb*32 + (e/8)*16 + (lane/16)*8 + e%8, N = lane%16.
// x rows are (b,t) flattened (m = b*T + t); output stored bf16 as [T][B][U].
// ---------------------------------------------------------------------------
__global__ __launch_bounds__(256) void xp_gemm_kernel(
    const float* __restrict__ x, const unsigned short* __restrict__ Wxb,
    const float* __restrict__ bias, unsigned short* __restrict__ xp)
{
    __shared__ __align__(16) unsigned short WxRaw[D_ * 16];      // 16 KB
    __shared__ __align__(16) unsigned short WxL[(D_ / 32) * 32 * 16]; // 16 KB

    const int lane  = threadIdx.x & 31;
    const int wave  = threadIdx.x >> 5;
    const int m0    = (blockIdx.y * 8 + wave) * 16;
    const int u0    = blockIdx.x * 16;
    const int khalf = lane >> 4;                 // 0: lanes 0-15, 1: lanes 16-31
    const int mrow  = m0 + (lane & 15);
    const int ucol  = u0 + (lane & 15);

    // --- TDM: stage raw Wx[:, u0:u0+16] tile into LDS -----------------------
    if (threadIdx.x < 32) {
        tdm_load_col_tile((unsigned)(uintptr_t)&WxRaw[0], Wxb + u0, D_);
        __builtin_amdgcn_s_wait_tensorcnt(0);
    }
    __syncthreads();
    // --- one-time swizzle into fragment-ready layout ------------------------
    for (int i = threadIdx.x; i < (D_ / 32) * 32 * 16; i += 256) {
        const int e  = i & 15;
        const int ln = (i >> 4) & 31;
        const int kb = i >> 9;
        const int k  = kb * 32 + ((e >> 3) << 4) + ((ln >> 4) << 3) + (e & 7);
        WxL[i] = WxRaw[k * 16 + (ln & 15)];
    }
    __syncthreads();

    const float* xrow = x + (size_t)mrow * D_;
    const float  bval = bias[ucol];

    v8f acc = {};
    for (int k0 = 0; k0 < D_; k0 += 32) {
        if (k0 + 32 < D_) __builtin_prefetch(xrow + k0 + 32, 0, 3);
        FragU a, bfr;
        // A: K = (e/8)*16 + khalf*8 + e%8  -> two contiguous float4 pairs
#pragma unroll
        for (int kk = 0; kk < 2; ++kk) {
            const float4 f0 = *(const float4*)(xrow + k0 + kk * 16 + khalf * 8);
            const float4 f1 = *(const float4*)(xrow + k0 + kk * 16 + khalf * 8 + 4);
            a.bh[kk * 8 + 0] = (__bf16)f0.x; a.bh[kk * 8 + 1] = (__bf16)f0.y;
            a.bh[kk * 8 + 2] = (__bf16)f0.z; a.bh[kk * 8 + 3] = (__bf16)f0.w;
            a.bh[kk * 8 + 4] = (__bf16)f1.x; a.bh[kk * 8 + 5] = (__bf16)f1.y;
            a.bh[kk * 8 + 6] = (__bf16)f1.z; a.bh[kk * 8 + 7] = (__bf16)f1.w;
        }
        // B fragment: one 32-byte LDS read
        const uint4* lp = (const uint4*)&WxL[((k0 >> 5) * 32 + lane) << 4];
        bfr.q[0] = lp[0];
        bfr.q[1] = lp[1];

        acc = __builtin_amdgcn_wmma_f32_16x16x32_bf16(false, a.v, false, bfr.v,
                                                      (short)0, acc, false, false);
    }
    // C/D layout: row M = khalf*8 + r, col N = lane%16
#pragma unroll
    for (int r = 0; r < 8; ++r) {
        const int m  = m0 + khalf * 8 + r;
        const int bb = m >> 9;            // m = b*T + t, T = 512
        const int tt = m & 511;
        xp[((size_t)tt * B_ + bb) * U_ + ucol] = f2bf(acc[r] + bval);
    }
}

// ---------------------------------------------------------------------------
// Persistent recurrence. 64 WGs x 128 threads (4 waves). WG w owns output
// columns [w*16, w*16+16); its 1024x16 slice of Wh is TDM-loaded into LDS
// raw (32 KB) then swizzled once into B-fragment order (32 KB), resident for
// all 512 steps. Each wave computes one 16x16 tile of h (B=64 -> 4 tiles).
// Grid barrier between timesteps via agent-scope atomics.
// ---------------------------------------------------------------------------
__global__ __launch_bounds__(128) void rnn_persist_kernel(
    const unsigned short* __restrict__ xp,   // [T][B][U] bf16
    const unsigned short* __restrict__ Whb,  // [U][U]    bf16
    unsigned short* __restrict__ h0,         // [B][U]    bf16 (zeroed)
    unsigned short* __restrict__ h1,         // [B][U]    bf16
    float* __restrict__ out,                 // [B][U]    f32
    unsigned* __restrict__ ctr)              // [0]=count, [1]=generation
{
    __shared__ __align__(16) unsigned short WhRaw[U_ * 16];          // 32 KB
    __shared__ __align__(16) unsigned short WhL[(U_ / 32) * 32 * 16]; // 32 KB

    const int lane  = threadIdx.x & 31;
    const int wave  = threadIdx.x >> 5;
    const int u0    = blockIdx.x * 16;
    const int m0    = wave * 16;
    const int khalf = lane >> 4;
    const int ucol  = u0 + (lane & 15);

    // --- TDM: stage raw Wh[:, u0:u0+16] tile into LDS -----------------------
    if (threadIdx.x < 32) {
        tdm_load_col_tile((unsigned)(uintptr_t)&WhRaw[0], Whb + u0, U_);
        __builtin_amdgcn_s_wait_tensorcnt(0);
    }
    __syncthreads();
    // --- one-time swizzle into fragment-ready layout ------------------------
    for (int i = threadIdx.x; i < (U_ / 32) * 32 * 16; i += 128) {
        const int e  = i & 15;
        const int ln = (i >> 4) & 31;
        const int kb = i >> 9;
        const int k  = kb * 32 + ((e >> 3) << 4) + ((ln >> 4) << 3) + (e & 7);
        WhL[i] = WhRaw[k * 16 + (ln & 15)];
    }
    __syncthreads();

    unsigned short* hbuf[2] = { h0, h1 };

    for (int t = 0; t < T_; ++t) {
        const unsigned short* hc = hbuf[t & 1];
        unsigned short*       hn = hbuf[(t + 1) & 1];

        if (t + 1 < T_)
            __builtin_prefetch(xp + (size_t)(t + 1) * (B_ * U_) +
                               (size_t)(m0 + (khalf << 3)) * U_ + ucol, 0, 0);

        v8f acc = {};
        const unsigned short* ha =
            hc + (size_t)(m0 + (lane & 15)) * U_ + (khalf << 3);
#pragma unroll 4
        for (int kb = 0; kb < 32; ++kb) {
            FragU a, bfr;
            a.q[0] = *(const uint4*)(ha + kb * 32);          // K = kb*32 + khalf*8 + 0..7
            a.q[1] = *(const uint4*)(ha + kb * 32 + 16);     // K = +16
            const uint4* lp = (const uint4*)&WhL[(kb * 32 + lane) << 4];
            bfr.q[0] = lp[0];
            bfr.q[1] = lp[1];
            acc = __builtin_amdgcn_wmma_f32_16x16x32_bf16(false, a.v, false, bfr.v,
                                                          (short)0, acc, false, false);
        }

        const unsigned short* xpt = xp + (size_t)t * (B_ * U_);
#pragma unroll
        for (int r = 0; r < 8; ++r) {
            const int   bb = m0 + (khalf << 3) + r;
            const float v  = tanhf(acc[r] + bf2f(xpt[(size_t)bb * U_ + ucol]));
            hn[(size_t)bb * U_ + ucol] = f2bf(v);
            if (t == T_ - 1) out[(size_t)bb * U_ + ucol] = v;
        }

        // ---- grid-wide barrier (release h stores, acquire for next step) ----
        __threadfence();
        __syncthreads();
        if (threadIdx.x == 0) {
            const unsigned g = __hip_atomic_load(&ctr[1], __ATOMIC_RELAXED,
                                                 __HIP_MEMORY_SCOPE_AGENT);
            const unsigned arrived = __hip_atomic_fetch_add(&ctr[0], 1u,
                                                            __ATOMIC_ACQ_REL,
                                                            __HIP_MEMORY_SCOPE_AGENT);
            if (arrived == NWG_RNN - 1) {
                __hip_atomic_store(&ctr[0], 0u, __ATOMIC_RELAXED,
                                   __HIP_MEMORY_SCOPE_AGENT);
                __hip_atomic_store(&ctr[1], g + 1u, __ATOMIC_RELEASE,
                                   __HIP_MEMORY_SCOPE_AGENT);
            } else {
                while (__hip_atomic_load(&ctr[1], __ATOMIC_ACQUIRE,
                                         __HIP_MEMORY_SCOPE_AGENT) == g) {
                    __builtin_amdgcn_s_sleep(1);
                }
            }
        }
        __syncthreads();
        __threadfence();
    }
}

// ---------------------------------------------------------------------------
extern "C" void kernel_launch(void* const* d_in, const int* in_sizes, int n_in,
                              void* d_out, int out_size, void* d_ws, size_t ws_size,
                              hipStream_t stream) {
    const float* x  = (const float*)d_in[0];   // [B,T,D]
    const float* Wx = (const float*)d_in[1];   // [D,U]
    const float* Wh = (const float*)d_in[2];   // [U,U]
    const float* bv = (const float*)d_in[3];   // [U]
    float* out = (float*)d_out;                // [B,U]

    // Workspace layout (bf16 unless noted):
    unsigned short* xp  = (unsigned short*)d_ws;        // [T][B][U]  64 MB
    unsigned short* Wxb = xp  + (size_t)T_ * B_ * U_;   // [D][U]      1 MB
    unsigned short* Whb = Wxb + (size_t)D_ * U_;        // [U][U]      2 MB
    unsigned short* h0  = Whb + (size_t)U_ * U_;        // [B][U]    128 KB
    unsigned short* h1  = h0  + (size_t)B_ * U_;        // [B][U]    128 KB
    unsigned*       ctr = (unsigned*)(h1 + (size_t)B_ * U_);  // 2 x u32

    rnn_init_kernel<<<128, 256, 0, stream>>>((unsigned*)h0, ctr);
    cvt_bf16_kernel<<<512, 256, 0, stream>>>(Wx, Wxb, D_ * U_);
    cvt_bf16_kernel<<<1024, 256, 0, stream>>>(Wh, Whb, U_ * U_);

    dim3 g1(U_ / 16, (B_ * T_) / (16 * 8));             // (64, 256) blocks
    xp_gemm_kernel<<<g1, 256, 0, stream>>>(x, Wxb, bv, xp);

    rnn_persist_kernel<<<NWG_RNN, 128, 0, stream>>>(xp, Whb, h0, h1, out, ctr);
}
